// IntegerLinear_1340029796554
// MI455X (gfx1250) — 8-bit matrix operand order repaired and checked by reading, compiled, not run on hardware
//
#include <hip/hip_runtime.h>


typedef int v8i __attribute__((ext_vector_type(8)));

#define B_ROWS 8192
#define IN_FEAT 4096
#define OUT_FEAT 4096
#define KSTEPS (IN_FEAT / 64)

#define OUT_SCALE 7.84313725490196e-05f
#define IN_SCALE_F (2.0f / 255.0f)

__global__ __launch_bounds__(256) void quant_x_kernel(const float* __restrict__ x,
                                                      unsigned int* __restrict__ xq) {
  const size_t i = (size_t)blockIdx.x * 256 + threadIdx.x;
  const float4 v = ((const float4*)x)[i];
  float q0 = fminf(fmaxf(rintf(v.x / IN_SCALE_F + 127.0f), 0.0f), 255.0f);
  float q1 = fminf(fmaxf(rintf(v.y / IN_SCALE_F + 127.0f), 0.0f), 255.0f);
  float q2 = fminf(fmaxf(rintf(v.z / IN_SCALE_F + 127.0f), 0.0f), 255.0f);
  float q3 = fminf(fmaxf(rintf(v.w / IN_SCALE_F + 127.0f), 0.0f), 255.0f);
  unsigned int b = ((unsigned int)q0) | (((unsigned int)q1) << 8) |
                   (((unsigned int)q2) << 16) | (((unsigned int)q3) << 24);
  xq[i] = b;
}

__global__ __launch_bounds__(256) void pack_w_kernel(const int* __restrict__ w,
                                                     unsigned int* __restrict__ wp) {
  const size_t i = (size_t)blockIdx.x * 256 + threadIdx.x;
  const int4 v = ((const int4*)w)[i];
  unsigned int b = ((unsigned int)v.x & 0xFFu) | (((unsigned int)v.y & 0xFFu) << 8) |
                   (((unsigned int)v.z & 0xFFu) << 16) | (((unsigned int)v.w & 0xFFu) << 24);
  wp[i] = b;
}

__global__ __launch_bounds__(256) void corr_kernel(const int* __restrict__ w,
                                                   const int* __restrict__ bq,
                                                   float* __restrict__ corr) {
  __shared__ int red[256];
  const int n = blockIdx.x;
  const int* row = w + (size_t)n * IN_FEAT;
  int s = 0;
  for (int k = threadIdx.x; k < IN_FEAT; k += 256) s += row[k];
  red[threadIdx.x] = s;
  __syncthreads();
  for (int off = 128; off > 0; off >>= 1) {
    if ((int)threadIdx.x < off) red[threadIdx.x] += red[threadIdx.x + off];
    __syncthreads();
  }
  if (threadIdx.x == 0) {
    corr[n] = (float)bq[n] * 0.01f - 127.0f * (float)red[0] * OUT_SCALE;
  }
}

__global__ __launch_bounds__(256) void gemm_iu8_kernel(const unsigned char* __restrict__ aq,
                                                       const unsigned char* __restrict__ wp,
                                                       const float* __restrict__ corr,
                                                       float* __restrict__ out) {
  __shared__ __align__(16) unsigned char As[2][256 * 64];
  __shared__ __align__(16) unsigned char Bs[2][128 * 64];

  const int tid = threadIdx.x;
  const int lane = tid & 31;
  const int wave = tid >> 5;
  const int wm = wave & 3;
  const int wn = wave >> 2;
  const int bm = blockIdx.y * 256;
  const int bn = blockIdx.x * 128;

  const unsigned char* gA = aq + (size_t)(bm + tid) * IN_FEAT;
  const unsigned char* gB = wp + (size_t)(bn + (tid >> 1)) * IN_FEAT + (size_t)(tid & 1) * 32;

  v8i acc[4][4];
#pragma unroll
  for (int i = 0; i < 4; ++i)
#pragma unroll
    for (int j = 0; j < 4; ++j)
#pragma unroll
      for (int e = 0; e < 8; ++e) acc[i][j][e] = 0;

  int4 ra[4];
  int4 rb[2];

#pragma unroll
  for (int c = 0; c < 4; ++c) ra[c] = *(const int4*)(gA + c * 16);
#pragma unroll
  for (int c = 0; c < 2; ++c) rb[c] = *(const int4*)(gB + c * 16);
  {
    int4* sa = (int4*)(As[0] + (size_t)tid * 64);
#pragma unroll
    for (int c = 0; c < 4; ++c) sa[c] = ra[c];
    int4* sb = (int4*)(Bs[0] + (size_t)(tid >> 1) * 64 + (size_t)(tid & 1) * 32);
#pragma unroll
    for (int c = 0; c < 2; ++c) sb[c] = rb[c];
  }
  __syncthreads();

  for (int s = 0; s < KSTEPS; ++s) {
    const int buf = s & 1;

    if (s + 1 < KSTEPS) {
      const size_t kb = (size_t)(s + 1) * 64;
#pragma unroll
      for (int c = 0; c < 4; ++c) ra[c] = *(const int4*)(gA + kb + c * 16);
#pragma unroll
      for (int c = 0; c < 2; ++c) rb[c] = *(const int4*)(gB + kb + c * 16);
    }

    v8i af[4];
#pragma unroll
    for (int i = 0; i < 4; ++i) {
      const unsigned char* pa =
          As[buf] + (size_t)(wm * 64 + i * 16 + (lane & 15)) * 64 + ((lane >> 4) << 4);
      uint2 d0 = *(const uint2*)(pa);
      uint2 d1 = *(const uint2*)(pa + 8);
      uint2 d2 = *(const uint2*)(pa + 32);
      uint2 d3 = *(const uint2*)(pa + 40);
      v8i f = {(int)d0.x, (int)d0.y, (int)d1.x, (int)d1.y,
               (int)d2.x, (int)d2.y, (int)d3.x, (int)d3.y};
      af[i] = f;
    }

#pragma unroll
    for (int j = 0; j < 4; ++j) {
      const unsigned char* pb =
          Bs[buf] + (size_t)(wn * 64 + j * 16 + (lane & 15)) * 64 + ((lane >> 4) << 4);
      int4 lo = *(const int4*)(pb);
      int4 hi = *(const int4*)(pb + 32);
      v8i bf = {lo.x, lo.y, lo.z, lo.w, hi.x, hi.y, hi.z, hi.w};
#pragma unroll
      for (int i = 0; i < 4; ++i) {
        acc[i][j] = __builtin_amdgcn_wmma_i32_16x16x64_iu8(
            false, af[i], true, bf, acc[i][j], false, false);
      }
    }

    if (s + 1 < KSTEPS) {
      int4* sa = (int4*)(As[buf ^ 1] + (size_t)tid * 64);
#pragma unroll
      for (int c = 0; c < 4; ++c) sa[c] = ra[c];
      int4* sb = (int4*)(Bs[buf ^ 1] + (size_t)(tid >> 1) * 64 + (size_t)(tid & 1) * 32);
#pragma unroll
      for (int c = 0; c < 2; ++c) sb[c] = rb[c];
    }
    __syncthreads();
  }

  const int colq = lane & 15;
  const int rsel = (lane >> 4) << 3;
#pragma unroll
  for (int j = 0; j < 4; ++j) {
    const int col = bn + wn * 64 + j * 16 + colq;
    const float cr = corr[col];
#pragma unroll
    for (int i = 0; i < 4; ++i) {
      const int row0 = bm + wm * 64 + i * 16 + rsel;
#pragma unroll
      for (int v = 0; v < 8; ++v) {
        out[(size_t)(row0 + v) * OUT_FEAT + col] = (float)acc[i][j][v] * OUT_SCALE + cr;
      }
    }
  }
}

extern "C" void kernel_launch(void* const* d_in, const int* in_sizes, int n_in,
                              void* d_out, int out_size, void* d_ws, size_t ws_size,
                              hipStream_t stream) {
  (void)in_sizes; (void)n_in; (void)out_size; (void)ws_size;

  const float* x = (const float*)d_in[0];
  const int* wq = (const int*)d_in[1];
  const int* bq = (const int*)d_in[2];
  float* out = (float*)d_out;

  unsigned char* ws = (unsigned char*)d_ws;
  const size_t A_BYTES = (size_t)B_ROWS * IN_FEAT;
  const size_t W_BYTES = (size_t)OUT_FEAT * IN_FEAT;
  unsigned int* xq = (unsigned int*)ws;
  unsigned int* wp = (unsigned int*)(ws + A_BYTES);
  float* corr = (float*)(ws + A_BYTES + W_BYTES);

  quant_x_kernel<<<(B_ROWS * (size_t)IN_FEAT) / 4 / 256, 256, 0, stream>>>(x, xq);
  pack_w_kernel<<<(OUT_FEAT * (size_t)IN_FEAT) / 4 / 256, 256, 0, stream>>>(wq, wp);
  corr_kernel<<<OUT_FEAT, 256, 0, stream>>>(wq, bq, corr);

  dim3 grid(OUT_FEAT / 128, B_ROWS / 256);
  gemm_iu8_kernel<<<grid, 256, 0, stream>>>((const unsigned char*)xq,
                                            (const unsigned char*)wp, corr, out);
}
